// BiLSTM_CRF_76450417868834
// MI455X (gfx1250) — matrix operand order repaired and checked by reading, not compiled, not run on hardware
//
#include <hip/hip_runtime.h>
#include <hip/hip_bf16.h>

#define DEVINL __device__ __forceinline__

typedef __attribute__((ext_vector_type(16))) __bf16 v16bf;
typedef __attribute__((ext_vector_type(8)))  float  v8f;

typedef __attribute__((ext_vector_type(4))) unsigned int tdm_u32x4;
typedef __attribute__((ext_vector_type(8))) int          tdm_i32x8;
typedef __attribute__((ext_vector_type(4))) int          tdm_i32x4;

#if defined(__has_builtin)
#if __has_builtin(__builtin_amdgcn_tensor_load_to_lds)
#define HAVE_TDM 1
#endif
#endif
#ifndef HAVE_TDM
#define HAVE_TDM 0
#endif

union Frag {
  v16bf v;
  unsigned int   w[8];
  unsigned short h[16];
  uint4          q[2];
};

constexpr int kT  = 4096;
constexpr int kE  = 256;
constexpr int kH  = 512;
constexpr int kG  = 4 * kH;   // 2048 gate rows
constexpr int kNT = 10;       // tags
constexpr int kSTART = 8;
constexpr int kSTOP  = 9;
constexpr float kNEG = -10000.0f;
constexpr int kSlices = 16;   // workgroups per direction in the recurrent kernel

DEVINL unsigned short f2bf(float f) {
  unsigned u = __builtin_bit_cast(unsigned, f);
  u += 0x7FFFu + ((u >> 16) & 1u);          // round-to-nearest-even
  return (unsigned short)(u >> 16);
}

DEVINL float sigmoidf_(float x) { return 1.0f / (1.0f + __expf(-x)); }

// ---- WMMA fragment loaders (CDNA5 wave32 layouts, CDNA5 ISA) ----
// A fragment: 16x32 bf16 tile at (row0,k0) of row-major M (ld elems).
// Per lane the 8 K-pairs are two contiguous 16B runs -> two b128 loads.
DEVINL void load_a_bf16(Frag& a, const unsigned short* M, int ld,
                        int row0, int k0, int lane) {
  const int m = row0 + (lane & 15);
  const int khalf = ((lane >> 4) & 1) * 16;
  const unsigned short* p = M + (size_t)m * ld + k0 + khalf;
  a.q[0] = *(const uint4*)p;
  a.q[1] = *(const uint4*)(p + 8);
}

// B fragment: 32x16 with B[k][n] = W[n][k] (W row-major, ld elems).
DEVINL void load_b_rowmajorT(Frag& b, const unsigned short* W, int ld,
                             int n0, int k0, int lane) {
  const int n = n0 + (lane & 15);
  const int kh = ((lane >> 4) & 1) * 16;
  const unsigned short* p = W + (size_t)n * ld + k0 + kh;
  b.q[0] = *(const uint4*)p;
  b.q[1] = *(const uint4*)(p + 8);
}

// Mat-vec B fragment: only column n==0 carries h[k0..k0+31] (bf16 in LDS).
DEVINL void load_b_vec(Frag& b, const unsigned short* hbf, int k0, int lane) {
  const int kh = ((lane >> 4) & 1) * 16;
  uint4 z = {0u, 0u, 0u, 0u};
  b.q[0] = z; b.q[1] = z;
  if ((lane & 15) == 0) {
    b.q[0] = *(const uint4*)(hbf + k0 + kh);
    b.q[1] = *(const uint4*)(hbf + k0 + kh + 8);
  }
}

// sense-reversing inter-workgroup barrier (per direction, 16 blocks)
DEVINL void grid_barrier(int* cnt, int* gen, int nblk) {
  __syncthreads();
  if (threadIdx.x == 0) {
    __threadfence();
    int g = __hip_atomic_load(gen, __ATOMIC_RELAXED, __HIP_MEMORY_SCOPE_AGENT);
    if (__hip_atomic_fetch_add(cnt, 1, __ATOMIC_ACQ_REL, __HIP_MEMORY_SCOPE_AGENT) == nblk - 1) {
      __hip_atomic_store(cnt, 0, __ATOMIC_RELAXED, __HIP_MEMORY_SCOPE_AGENT);
      __hip_atomic_fetch_add(gen, 1, __ATOMIC_RELEASE, __HIP_MEMORY_SCOPE_AGENT);
    } else {
      while (__hip_atomic_load(gen, __ATOMIC_ACQUIRE, __HIP_MEMORY_SCOPE_AGENT) == g)
        __builtin_amdgcn_s_sleep(1);
    }
  }
  __syncthreads();
}

// ---- Kernel 0: gather embeddings, weights->bf16, init h-broadcast + barrier -
__global__ void prep_kernel(const int* __restrict__ widx, const float* __restrict__ emb,
                            const float* __restrict__ wihF, const float* __restrict__ wihB,
                            const float* __restrict__ whhF, const float* __restrict__ whhB,
                            const float* __restrict__ h0,
                            unsigned short* __restrict__ xbf,
                            unsigned short* __restrict__ wih,
                            unsigned short* __restrict__ whh,
                            unsigned short* __restrict__ hglob,
                            int* __restrict__ gbar) {
  const int n0 = kT * kE;
  const int n1 = kG * kE;
  const int n2 = kG * kH;
  const int total = n0 + 2 * n1 + 2 * n2;
  const int gid = blockIdx.x * blockDim.x + threadIdx.x;
  for (int i = gid; i < total; i += gridDim.x * blockDim.x) {
    if (i < n0) {
      const int t = i / kE, e = i % kE;
      xbf[i] = f2bf(emb[(size_t)widx[t] * kE + e]);
    } else if (i < n0 + 2 * n1) {
      const int j = i - n0;
      const float* src = (j < n1) ? wihF : wihB;
      wih[j] = f2bf(src[(j < n1) ? j : j - n1]);
    } else {
      const int j = i - n0 - 2 * n1;
      const float* src = (j < n2) ? whhF : whhB;
      whh[j] = f2bf(src[(j < n2) ? j : j - n2]);
    }
  }
  if (gid < 4) gbar[gid] = 0;                      // barrier count/gen, both dirs
  if (gid < 2 * kH) {                              // h0 -> broadcast buffer 0
    const int d = gid >> 9;
    hglob[d * 2 * kH + (gid & (kH - 1))] = f2bf(h0[gid]);
  }
}

// ---- Kernel 1: Gpre[d][t][g] = b[d][g] + sum_e x[t][e]*Wih[d][g][e] --------
// 1x4 register blocking: one A fragment feeds 4 B fragments / 4 accumulators
// (4x less A traffic from L2, 8 back-to-back WMMAs per unrolled body).
__global__ void gpre_kernel(const unsigned short* __restrict__ xbf,
                            const unsigned short* __restrict__ wih,
                            const float* __restrict__ bF, const float* __restrict__ bB,
                            float* __restrict__ gpre) {
  const int wid  = blockIdx.x * (blockDim.x >> 5) + (threadIdx.x >> 5);
  const int lane = threadIdx.x & 31;
  const int nt4 = wid & 31;           // group of 4 consecutive N tiles
  const int mt  = (wid >> 5) & 255;
  const int dir = wid >> 13;          // 2 * 256 * 32 waves

  const unsigned short* W = wih + (size_t)dir * kG * kE;
  const float* bias = dir ? bB : bF;
  const int n0 = nt4 * 64;
  __builtin_prefetch(W + (size_t)(n0 + (lane & 15)) * kE, 0, 0);

  v8f acc0 = {}, acc1 = {}, acc2 = {}, acc3 = {};
  Frag a, b0, b1, b2, b3;
#pragma unroll 2
  for (int kk = 0; kk < kE; kk += 32) {
    load_a_bf16(a, xbf, kE, mt * 16, kk, lane);
    load_b_rowmajorT(b0, W, kE, n0 +  0, kk, lane);
    load_b_rowmajorT(b1, W, kE, n0 + 16, kk, lane);
    load_b_rowmajorT(b2, W, kE, n0 + 32, kk, lane);
    load_b_rowmajorT(b3, W, kE, n0 + 48, kk, lane);
    acc0 = __builtin_amdgcn_wmma_f32_16x16x32_bf16(false, a.v, false, b0.v,
                                                   (short)0, acc0, false, false);
    acc1 = __builtin_amdgcn_wmma_f32_16x16x32_bf16(false, a.v, false, b1.v,
                                                   (short)0, acc1, false, false);
    acc2 = __builtin_amdgcn_wmma_f32_16x16x32_bf16(false, a.v, false, b2.v,
                                                   (short)0, acc2, false, false);
    acc3 = __builtin_amdgcn_wmma_f32_16x16x32_bf16(false, a.v, false, b3.v,
                                                   (short)0, acc3, false, false);
  }
  const int half8 = ((lane >> 4) & 1) * 8;
  const int row   = mt * 16 + half8;
  float* out = gpre + (size_t)dir * kT * kG;
  const v8f* accs[4] = {&acc0, &acc1, &acc2, &acc3};
#pragma unroll
  for (int tt = 0; tt < 4; ++tt) {
    const int col  = n0 + tt * 16 + (lane & 15);
    const float bv = bias[col];
    const v8f& ac  = *accs[tt];
#pragma unroll
    for (int r = 0; r < 8; ++r)
      out[(size_t)(row + r) * kG + col] = ac[r] + bv;
  }
}

// ---- Kernel 2: persistent-RNN recurrence ----------------------------------
// 32 blocks = 2 dirs x 16 slices; each block keeps a 128KB bf16 slice of Whh
// resident in LDS (TDM-staged), owns h[w*32 .. w*32+32) and its i/f/g/o gate
// rows. Per step: LDS-resident WMMA mat-vec, local gate update, 1KB h
// broadcast via L2 with one inter-workgroup barrier (double-buffered h).
__global__ void __launch_bounds__(256)
lstm_kernel(const float* __restrict__ gpre, const unsigned short* __restrict__ whh,
            const float* __restrict__ c0, float* __restrict__ hout,
            unsigned short* __restrict__ hglob, int* __restrict__ gbar) {
  extern __shared__ unsigned char dynsm[];
  unsigned short* wsl = (unsigned short*)dynsm;     // [8 tiles][16][512] bf16 = 128 KB
  __shared__ float gates_lds[4 * 32];               // i/f/g/o for this block's 32 h
  __shared__ __align__(16) unsigned short hbf[kH];  // full h, bf16

  const int dir  = blockIdx.x >> 4;
  const int w    = blockIdx.x & 15;                 // h-slice index
  const int tid  = threadIdx.x;
  const int lane = tid & 31;
  const int wave = tid >> 5;                        // 8 waves = 8 gate tiles
  const int gate = wave >> 1;
  const int hh   = wave & 1;
  const int row0g = gate * kH + w * 32 + hh * 16;   // global gate-row of this tile

  int* cnt = gbar + dir * 2;
  int* gen = gbar + dir * 2 + 1;
  const float* gp = gpre + (size_t)dir * kT * kG;
  float* ho = hout + (size_t)dir * kT * kH;
  unsigned short* hg = hglob + dir * 2 * kH;        // [2][512] double buffer

  // ---- stage this wave's 16x512 bf16 tile of Whh into LDS ----
  {
    const unsigned short* gsrc = whh + ((size_t)dir * kG + row0g) * kH;
    unsigned short* ldst = wsl + (size_t)wave * 16 * kH;
#if HAVE_TDM
    const unsigned long long ga = (unsigned long long)(size_t)gsrc;
    tdm_u32x4 g0;
    g0[0] = 1u;                                        // count=1 user D#
    g0[1] = (unsigned int)(size_t)ldst;                // lds_addr
    g0[2] = (unsigned int)(ga & 0xffffffffull);        // global_addr lo
    g0[3] = (unsigned int)((ga >> 32) & 0x01ffffffull) | 0x80000000u; // hi | type=2
    tdm_i32x8 g1;
    g1[0] = 0x00010000;          // data_size=2B, no multicast/pad/iterate
    g1[1] = (int)(512u << 16);   // tensor_dim0 = 512
    g1[2] = (int)(16u << 16);    // tensor_dim1 = 16
    g1[3] = (int)(512u << 16);   // tile_dim0 = 512
    g1[4] = 16;                  // tile_dim1 = 16
    g1[5] = 512;                 // tensor_dim0_stride = 512
    g1[6] = 0; g1[7] = 0;
    tdm_i32x4 gz = {0, 0, 0, 0};
#if __clang_major__ >= 23
    tdm_i32x8 gz8 = {0, 0, 0, 0, 0, 0, 0, 0};
    __builtin_amdgcn_tensor_load_to_lds(g0, g1, gz, gz, gz8, 0);
#else
    __builtin_amdgcn_tensor_load_to_lds(g0, g1, gz, gz, 0);
#endif
    __builtin_amdgcn_s_wait_tensorcnt(0);
#else
    for (int i = lane; i < 16 * kH / 8; i += 32)
      ((uint4*)ldst)[i] = ((const uint4*)gsrc)[i];
#endif
  }

  float c_reg = 0.0f;
  if (tid < 32) c_reg = c0[dir * kH + w * 32 + tid];
  __syncthreads();

  const unsigned short* wtile = wsl + (size_t)wave * 16 * kH;

  for (int step = 0; step < kT; ++step) {
    const int t = dir ? (kT - 1 - step) : step;

    // pull current h (1 KB) from the L2-resident broadcast buffer into LDS
    if (tid < kH / 2)
      ((unsigned int*)hbf)[tid] = ((const unsigned int*)(hg + (step & 1) * kH))[tid];
    __syncthreads();

    // 16x512 . h mat-vec tile on the matrix pipe, operands in LDS
    v8f acc = {};
    Frag a, b;
#pragma unroll 4
    for (int kk = 0; kk < kH; kk += 32) {
      load_a_bf16(a, wtile, kH, 0, kk, lane);
      load_b_vec(b, hbf, kk, lane);
      acc = __builtin_amdgcn_wmma_f32_16x16x32_bf16(false, a.v, false, b.v,
                                                    (short)0, acc, false, false);
    }
    if ((lane & 15) == 0) {        // col 0 of D: lanes 0 (M=r) and 16 (M=8+r)
      const int half8 = ((lane >> 4) & 1) * 8;
      const float* gprow = gp + (size_t)t * kG + row0g;
#pragma unroll
      for (int r = 0; r < 8; ++r)
        gates_lds[wave * 16 + half8 + r] = acc[r] + gprow[half8 + r];
    }
    __syncthreads();

    if (tid < 32) {                // this block's 32 hidden units
      const float gi = gates_lds[tid];
      const float gf = gates_lds[32 + tid];
      const float gg = gates_lds[64 + tid];
      const float go = gates_lds[96 + tid];
      c_reg = sigmoidf_(gf) * c_reg + sigmoidf_(gi) * tanhf(gg);
      const float h = sigmoidf_(go) * tanhf(c_reg);
      const int hidx = w * 32 + tid;
      ho[(size_t)t * kH + hidx] = h;
      hg[((step + 1) & 1) * kH + hidx] = f2bf(h);
    }
    grid_barrier(cnt, gen, kSlices);
  }
}

// ---- Kernel 3: feats[t][n] = fcb[n] + cat(hf,hb)[t] . fcW[n] ---------------
__global__ void feats_kernel(const float* __restrict__ hout,
                             const float* __restrict__ fcW, const float* __restrict__ fcb,
                             float* __restrict__ feats) {
  const int wid  = blockIdx.x * (blockDim.x >> 5) + (threadIdx.x >> 5);  // 0..255
  const int lane = threadIdx.x & 31;
  const int mt   = wid;
  const float* hf = hout;
  const float* hb = hout + (size_t)kT * kH;

  v8f acc = {};
  Frag a, b;
  for (int kk = 0; kk < 2 * kH; kk += 32) {
    const int m = mt * 16 + (lane & 15);
    const int khalf = ((lane >> 4) & 1) * 16;
#pragma unroll
    for (int v = 0; v < 8; ++v) {
      const int k = kk + ((v & 4) ? 8 : 0) + khalf + (v & 3) * 2;  // always even
      const float* src = (k < kH) ? (hf + (size_t)m * kH + k)
                                  : (hb + (size_t)m * kH + (k - kH));
      a.h[2 * v]     = f2bf(src[0]);
      a.h[2 * v + 1] = f2bf(src[1]);
    }
    const int n  = lane & 15;
    const int kh = ((lane >> 4) & 1) * 16;
#pragma unroll
    for (int j = 0; j < 8; ++j) {
      const int k = kk + kh + 2 * j;
      unsigned short lo = 0, hi = 0;
      if (n < kNT) {
        lo = f2bf(fcW[(size_t)n * (2 * kH) + k]);
        hi = f2bf(fcW[(size_t)n * (2 * kH) + k + 1]);
      }
      b.h[2 * j] = lo;
      b.h[2 * j + 1] = hi;
    }
    acc = __builtin_amdgcn_wmma_f32_16x16x32_bf16(false, a.v, false, b.v,
                                                  (short)0, acc, false, false);
  }
  const int n     = lane & 15;
  const int half8 = ((lane >> 4) & 1) * 8;
  const float bv  = (n < kNT) ? fcb[n] : 0.0f;
#pragma unroll
  for (int r = 0; r < 8; ++r)
    feats[(size_t)(mt * 16 + r + half8) * 16 + n] = acc[r] + bv;
}

// ---- Kernel 4: gold-path score (parallel) + CRF forward scan (one wave) ----
__global__ void crf_kernel(const int* __restrict__ tidx, const float* __restrict__ trans,
                           const float* __restrict__ feats, float* __restrict__ out) {
  __shared__ float red[256];
  const int tid = threadIdx.x;

  float s = 0.0f;
  for (int t = tid; t < kT; t += 256) {
    const int tag  = tidx[t];
    const int prev = (t == 0) ? kSTART : tidx[t - 1];
    s += feats[(size_t)t * 16 + tag] + trans[tag * kNT + prev];
  }
  red[tid] = s;
  __syncthreads();
  for (int w = 128; w > 0; w >>= 1) {
    if (tid < w) red[tid] += red[tid + w];
    __syncthreads();
  }
  if (tid == 0) out[0] = red[0] + trans[kSTOP * kNT + tidx[kT - 1]];
  __syncthreads();

  if (tid < 32) {
    const int j = tid;
    float alpha = (j == kSTART) ? 0.0f : kNEG;
    float trow[kNT];
    if (j < kNT)
      for (int i = 0; i < kNT; ++i) trow[i] = trans[j * kNT + i];
    for (int t = 0; t < kT; ++t) {
      float av[kNT];
#pragma unroll
      for (int i = 0; i < kNT; ++i) av[i] = __shfl(alpha, i, 32);
      if (j < kNT) {
        float m = -3.0e38f;
#pragma unroll
        for (int i = 0; i < kNT; ++i) m = fmaxf(m, av[i] + trow[i]);
        float sum = 0.0f;
#pragma unroll
        for (int i = 0; i < kNT; ++i) sum += __expf(av[i] + trow[i] - m);
        alpha = m + __logf(sum) + feats[(size_t)t * 16 + j];
      }
    }
    float val = (j < kNT) ? alpha + trans[kSTOP * kNT + j] : -3.0e38f;
    float m = val;
#pragma unroll
    for (int o = 16; o > 0; o >>= 1) m = fmaxf(m, __shfl_xor(m, o, 32));
    float e = (j < kNT) ? __expf(val - m) : 0.0f;
#pragma unroll
    for (int o = 16; o > 0; o >>= 1) e += __shfl_xor(e, o, 32);
    if (j == 0) out[1] = m + __logf(e);
  }
}

extern "C" void kernel_launch(void* const* d_in, const int* in_sizes, int n_in,
                              void* d_out, int out_size, void* d_ws, size_t ws_size,
                              hipStream_t stream) {
  const int*   widx  = (const int*)d_in[0];
  const int*   tidx  = (const int*)d_in[1];
  const float* emb   = (const float*)d_in[2];
  const float* wihF  = (const float*)d_in[3];
  const float* whhF  = (const float*)d_in[4];
  const float* bF    = (const float*)d_in[5];
  const float* wihB  = (const float*)d_in[6];
  const float* whhB  = (const float*)d_in[7];
  const float* bB    = (const float*)d_in[8];
  const float* fcW   = (const float*)d_in[9];
  const float* fcb   = (const float*)d_in[10];
  const float* h0    = (const float*)d_in[11];
  const float* c0    = (const float*)d_in[12];
  const float* trans = (const float*)d_in[13];
  float* out = (float*)d_out;

  char* ws = (char*)d_ws;
  size_t off = 0;
  unsigned short* xbf = (unsigned short*)(ws + off); off += (size_t)kT * kE * 2;       // 2 MB
  unsigned short* wih = (unsigned short*)(ws + off); off += (size_t)2 * kG * kE * 2;   // 2 MB
  unsigned short* whh = (unsigned short*)(ws + off); off += (size_t)2 * kG * kH * 2;   // 4 MB
  float* gpre  = (float*)(ws + off); off += (size_t)2 * kT * kG * 4;                   // 64 MB
  float* hout  = (float*)(ws + off); off += (size_t)2 * kT * kH * 4;                   // 16 MB
  float* feats = (float*)(ws + off); off += (size_t)kT * 16 * 4;                       // 256 KB
  unsigned short* hglob = (unsigned short*)(ws + off); off += (size_t)2 * 2 * kH * 2;  // 4 KB
  int* gbar = (int*)(ws + off); off += 256;

  prep_kernel<<<4096, 256, 0, stream>>>(widx, emb, wihF, wihB, whhF, whhB, h0,
                                        xbf, wih, whh, hglob, gbar);
  gpre_kernel<<<2048, 256, 0, stream>>>(xbf, wih, bF, bB, gpre);  // 2*256*32 wave-groups
  lstm_kernel<<<2 * kSlices, 256, 8 * 16 * kH * sizeof(unsigned short), stream>>>(
      gpre, whh, c0, hout, hglob, gbar);                           // 128 KB LDS/block
  feats_kernel<<<32, 256, 0, stream>>>(hout, fcW, fcb, feats);
  crf_kernel<<<1, 256, 0, stream>>>(tidx, trans, feats, out);
}
